// EnhancedPretrainingBlock_14894946583243
// MI455X (gfx1250) — matrix operand order repaired and checked by reading, not compiled, not run on hardware
//
#include <hip/hip_runtime.h>
#include <hip/hip_bf16.h>

// ---------------- constants (fixed by reference) ----------------
constexpr int Dm  = 2048;   // model dim
constexpr int Hh  = 16;     // query heads
constexpr int KVh = 4;      // kv heads
constexpr int HDd = 128;    // head dim
constexpr int Ff  = 5632;   // ffn dim
constexpr int Ss  = 2048;   // sequence
constexpr int WINc = 512;
constexpr int NGc  = 64;
constexpr float EPSc = 1e-6f;

typedef __bf16 bf16_t;
typedef __attribute__((ext_vector_type(16))) __bf16 v16bf;
typedef __attribute__((ext_vector_type(8)))  __bf16 v8bf;
typedef __attribute__((ext_vector_type(8)))  float  v8f;

__device__ inline v16bf bfcat(v8bf lo, v8bf hi) {
  return __builtin_shufflevector(lo, hi, 0,1,2,3,4,5,6,7,8,9,10,11,12,13,14,15);
}

// A-matrix fragment (16x32 bf16): lane<16 holds K {0..7,16..23}, lane>=16 K {8..15,24..31}
__device__ inline v16bf load_fragA(const bf16_t* row, int lane) {
  int k = (lane & 16) ? 8 : 0;
  return bfcat(*(const v8bf*)(row + k), *(const v8bf*)(row + k + 16));
}
// B-matrix fragment from B^T row (N-row, K contiguous): lane<16 K {0..7,16..23}, lane>=16 K {8..15,24..31}
__device__ inline v16bf load_fragB(const bf16_t* row, int lane) {
  int k = (lane & 16) ? 8 : 0;
  return bfcat(*(const v8bf*)(row + k), *(const v8bf*)(row + k + 16));
}

// ---- CDNA5 async global->LDS copy (16B per lane), tracked by ASYNCcnt ----
__device__ inline void async_copy_b128(unsigned lds_off, const void* gptr) {
  asm volatile("global_load_async_to_lds_b128 %0, %1, off"
               :: "v"(lds_off), "v"((unsigned long long)(uintptr_t)gptr)
               : "memory");
}
__device__ inline void wait_async0() {
  asm volatile("s_wait_asynccnt 0" ::: "memory");
}
__device__ inline unsigned lds_offset(const void* p) {
  return (unsigned)(uintptr_t)p;   // flat->LDS uses addr[31:0] (ISA 10.2)
}

// ---------------- weight transpose + fp32->bf16 ----------------
// W [K,N] fp32 row-major  ->  WT [N,K] bf16 row-major
__global__ __launch_bounds__(256) void k_transpose_bf16(
    const float* __restrict__ W, bf16_t* __restrict__ WT, int K, int N) {
  __shared__ float tile[32][33];
  int n0 = blockIdx.x * 32, k0 = blockIdx.y * 32;
  int tx = threadIdx.x & 31, ty = threadIdx.x >> 5;   // 32x8
  #pragma unroll
  for (int i = ty; i < 32; i += 8)
    tile[i][tx] = W[(size_t)(k0 + i) * N + n0 + tx];
  __syncthreads();
  #pragma unroll
  for (int i = ty; i < 32; i += 8)
    WT[(size_t)(n0 + i) * K + k0 + tx] = (bf16_t)tile[tx][i];
}

// ---------------- RMSNorm -> bf16 ----------------
__global__ __launch_bounds__(256) void k_rmsnorm_bf16(
    const float* __restrict__ x, const float* __restrict__ g,
    bf16_t* __restrict__ out) {
  int row = blockIdx.x;
  const float* xr = x + (size_t)row * Dm;
  float s = 0.f;
  for (int i = threadIdx.x; i < Dm; i += 256) { float v = xr[i]; s += v * v; }
  #pragma unroll
  for (int m = 16; m >= 1; m >>= 1) s += __shfl_xor(s, m, 32);
  __shared__ float red[8];
  if ((threadIdx.x & 31) == 0) red[threadIdx.x >> 5] = s;
  __syncthreads();
  float tot = 0.f;
  #pragma unroll
  for (int i = 0; i < 8; ++i) tot += red[i];
  float inv = rsqrtf(tot / (float)Dm + EPSc);
  for (int i = threadIdx.x; i < Dm; i += 256)
    out[(size_t)row * Dm + i] = (bf16_t)(xr[i] * inv * g[i]);
}

// ---------------- bf16 WMMA GEMM: C = A[MxK] * B[KxN] (+Cin), BT is [N,K] ----------------
// 128x128 tile, 8 waves; each wave computes 64x32 (4 M-subtiles x 2 N-subtiles).
// Double-buffered LDS staged with CDNA5 async global->LDS copies.
__global__ __launch_bounds__(256) void k_gemm_bf16(
    const bf16_t* __restrict__ A, const bf16_t* __restrict__ BT,
    const float* __restrict__ Cin, float* __restrict__ Cout,
    int M, int N, int K) {
  __shared__ bf16_t As[2][128][32];
  __shared__ bf16_t Bs[2][128][32];
  int n0 = blockIdx.x * 128, m0 = blockIdx.y * 128;
  int tid = threadIdx.x, wave = tid >> 5, lane = tid & 31;
  int rr = lane & 15, half = lane >> 4;
  int wm = wave >> 2, wn = wave & 3;   // 2 x 4 wave grid

  v8f acc[4][2];
  #pragma unroll
  for (int mt = 0; mt < 4; ++mt)
    #pragma unroll
    for (int nt = 0; nt < 2; ++nt) acc[mt][nt] = v8f{};

  auto stage = [&](int buf, int k0) {
    #pragma unroll
    for (int p = 0; p < 2; ++p) {
      int idx = tid + 256 * p;          // 0..511 chunks of 16B (8KB per matrix)
      int r = idx >> 2;                 // 0..127
      int c = (idx & 3) * 8;            // bf16 elems: 0,8,16,24
      async_copy_b128(lds_offset(&As[buf][r][c]),
                      &A[(size_t)(m0 + r) * K + k0 + c]);
      async_copy_b128(lds_offset(&Bs[buf][r][c]),
                      &BT[(size_t)(n0 + r) * K + k0 + c]);
    }
  };

  int nsteps = K / 32;
  stage(0, 0);
  for (int s = 0; s < nsteps; ++s) {
    int cur = s & 1;
    wait_async0();          // this wave's tile landed in LDS
    __syncthreads();        // every wave's tile landed
    if (s + 1 < nsteps) stage(cur ^ 1, (s + 1) * 32);  // DMA next while computing
    v16bf afr[4], bfr[2];
    #pragma unroll
    for (int mt = 0; mt < 4; ++mt)
      afr[mt] = load_fragA(&As[cur][wm * 64 + mt * 16 + rr][0], lane);
    #pragma unroll
    for (int nt = 0; nt < 2; ++nt)
      bfr[nt] = load_fragB(&Bs[cur][wn * 32 + nt * 16 + rr][0], lane);
    #pragma unroll
    for (int mt = 0; mt < 4; ++mt)
      #pragma unroll
      for (int nt = 0; nt < 2; ++nt)
        acc[mt][nt] = __builtin_amdgcn_wmma_f32_16x16x32_bf16(
            false, afr[mt], false, bfr[nt], (short)0, acc[mt][nt], false, false);
  }
  #pragma unroll
  for (int mt = 0; mt < 4; ++mt)
    #pragma unroll
    for (int nt = 0; nt < 2; ++nt)
      #pragma unroll
      for (int r = 0; r < 8; ++r) {
        size_t idx = (size_t)(m0 + wm * 64 + mt * 16 + r + 8 * half) * N +
                     (n0 + wn * 32 + nt * 16 + rr);
        float v = acc[mt][nt][r];
        if (Cin != nullptr) v += Cin[idx];
        Cout[idx] = v;
      }
}

// ---------------- QK-norm + V transpose -> bf16 ----------------
__global__ __launch_bounds__(128) void k_qkv_prep(
    const float* __restrict__ qf, const float* __restrict__ kf,
    const float* __restrict__ vf, bf16_t* __restrict__ qb,
    bf16_t* __restrict__ kb, bf16_t* __restrict__ vT) {
  int s = blockIdx.x, head = blockIdx.y, d = threadIdx.x;
  __shared__ float red[4];
  float val;
  size_t idx;
  if (head < Hh) idx = ((size_t)s * Hh + head) * HDd + d;
  else           idx = ((size_t)s * KVh + (head - Hh)) * HDd + d;
  val = (head < Hh) ? qf[idx] : kf[idx];
  float ss = val * val;
  #pragma unroll
  for (int m = 16; m >= 1; m >>= 1) ss += __shfl_xor(ss, m, 32);
  if ((threadIdx.x & 31) == 0) red[threadIdx.x >> 5] = ss;
  __syncthreads();
  float tot = red[0] + red[1] + red[2] + red[3];
  float inv = rsqrtf(tot / (float)HDd + EPSc);
  if (head < Hh) {
    qb[idx] = (bf16_t)(val * inv);
  } else {
    int kv = head - Hh;
    kb[idx] = (bf16_t)(val * inv);
    vT[((size_t)kv * HDd + d) * Ss + s] = (bf16_t)vf[idx];   // [KV,HD,S]
  }
}

// ---------------- flash attention: 1 wave per (16-row tile, head) ----------------
__global__ __launch_bounds__(32) void k_attn(
    const bf16_t* __restrict__ qb,   // [S,H,HD]  qk-normalized
    const bf16_t* __restrict__ kb,   // [S,KV,HD] qk-normalized
    const bf16_t* __restrict__ vT,   // [KV,HD,S]
    bf16_t* __restrict__ ob) {       // [S,H*HD]
  int i0 = blockIdx.x * 16, h = blockIdx.y, kv = h / (Hh / KVh);
  int lane = threadIdx.x, rr = lane & 15, half = lane >> 4;
  __shared__ bf16_t Ps[16][32];

  v16bf qfrag[4];
  {
    const bf16_t* qrow = qb + ((size_t)(i0 + rr) * Hh + h) * HDd;
    #pragma unroll
    for (int c = 0; c < 4; ++c) qfrag[c] = load_fragA(qrow + c * 32, lane);
  }
  v8f oacc[8];
  #pragma unroll
  for (int i = 0; i < 8; ++i) oacc[i] = v8f{};
  float m_run[8], l_run[8];
  #pragma unroll
  for (int r = 0; r < 8; ++r) { m_run[r] = -1e30f; l_run[r] = 0.f; }

  const float scale = 0.08838834764831845f;  // 128^-0.5
  int jmax = i0 + 15;
  for (int j0 = 0; j0 <= jmax; j0 += 32) {
    // skip key tiles fully outside (window | global col | global row | causal)
    if (i0 >= NGc && j0 >= NGc && (i0 - (j0 + 31)) >= WINc) continue;
    // ---- scores: two 16x16 D tiles over 32 keys ----
    v8f sc[2] = {v8f{}, v8f{}};
    #pragma unroll
    for (int t = 0; t < 2; ++t) {
      const bf16_t* krow = kb + ((size_t)(j0 + t * 16 + rr) * KVh + kv) * HDd;
      #pragma unroll
      for (int c = 0; c < 4; ++c) {
        v16bf kfrag = load_fragB(krow + c * 32, lane);  // B^T row == k row
        sc[t] = __builtin_amdgcn_wmma_f32_16x16x32_bf16(
            false, qfrag[c], false, kfrag, (short)0, sc[t], false, false);
      }
    }
    // ---- mask + online softmax (rows r, r+8 per lane-half) ----
    #pragma unroll
    for (int r = 0; r < 8; ++r) {
      int irow = i0 + r + 8 * half;
      #pragma unroll
      for (int t = 0; t < 2; ++t) {
        int jcol = j0 + t * 16 + rr;
        float v = sc[t][r] * scale;
        bool ok = (jcol <= irow) &&
                  (((irow - jcol) < WINc) || (jcol < NGc) || (irow < NGc));
        sc[t][r] = ok ? v : -1e30f;
      }
      float mx = fmaxf(sc[0][r], sc[1][r]);
      #pragma unroll
      for (int m = 8; m >= 1; m >>= 1) mx = fmaxf(mx, __shfl_xor(mx, m, 32));
      float mnew = fmaxf(m_run[r], mx);
      float corr = __expf(m_run[r] - mnew);
      float p0 = __expf(sc[0][r] - mnew);
      float p1 = __expf(sc[1][r] - mnew);
      sc[0][r] = p0; sc[1][r] = p1;
      float rs = p0 + p1;
      #pragma unroll
      for (int m = 8; m >= 1; m >>= 1) rs += __shfl_xor(rs, m, 32);
      l_run[r] = l_run[r] * corr + rs;
      m_run[r] = mnew;
      #pragma unroll
      for (int dsub = 0; dsub < 8; ++dsub) oacc[dsub][r] *= corr;
    }
    // ---- re-layout P (D-tile) -> A-fragment through LDS ----
    #pragma unroll
    for (int t = 0; t < 2; ++t)
      #pragma unroll
      for (int r = 0; r < 8; ++r)
        Ps[r + 8 * half][t * 16 + rr] = (bf16_t)sc[t][r];
    __syncthreads();
    v16bf pfrag = load_fragA(&Ps[rr][0], lane);
    // ---- P(16x32) x V(32x128): 8 d-subtiles ----
    #pragma unroll
    for (int dsub = 0; dsub < 8; ++dsub) {
      const bf16_t* vrow = vT + ((size_t)kv * HDd + dsub * 16 + rr) * Ss + j0;
      v16bf vfrag = load_fragB(vrow, lane);
      oacc[dsub] = __builtin_amdgcn_wmma_f32_16x16x32_bf16(
          false, pfrag, false, vfrag, (short)0, oacc[dsub], false, false);
    }
    __syncthreads();
  }
  // ---- normalize + store o as bf16 [S, H*HD] ----
  float invl[8];
  #pragma unroll
  for (int r = 0; r < 8; ++r) invl[r] = 1.0f / l_run[r];
  #pragma unroll
  for (int dsub = 0; dsub < 8; ++dsub)
    #pragma unroll
    for (int r = 0; r < 8; ++r)
      ob[(size_t)(i0 + r + 8 * half) * (Hh * HDd) + h * HDd + dsub * 16 + rr] =
          (bf16_t)(oacc[dsub][r] * invl[r]);
}

// ---------------- SwiGLU elementwise -> bf16 ----------------
__global__ __launch_bounds__(256) void k_swiglu(
    const float* __restrict__ a, const float* __restrict__ b,
    bf16_t* __restrict__ g, size_t n) {
  size_t i = (size_t)blockIdx.x * 256 + threadIdx.x;
  if (i < n) {
    float x = a[i];
    float s = x * (1.0f / (1.0f + __expf(-x)));
    g[i] = (bf16_t)(s * b[i]);
  }
}

// ---------------- host orchestration ----------------
extern "C" void kernel_launch(void* const* d_in, const int* in_sizes, int n_in,
                              void* d_out, int out_size, void* d_ws, size_t ws_size,
                              hipStream_t stream) {
  const float* x  = (const float*)d_in[0];
  const float* ga = (const float*)d_in[1];
  const float* gf = (const float*)d_in[2];
  const float* wq = (const float*)d_in[3];
  const float* wk = (const float*)d_in[4];
  const float* wv = (const float*)d_in[5];
  const float* wo = (const float*)d_in[6];
  const float* w1 = (const float*)d_in[7];
  const float* w3 = (const float*)d_in[8];   // dict order: w1, w3, w2
  const float* w2 = (const float*)d_in[9];
  float* out = (float*)d_out;

  char* base = (char*)d_ws;
  size_t off = 0;
  auto alloc = [&](size_t bytes) -> void* {
    void* r = base + off;
    off = (off + bytes + 255) & ~(size_t)255;
    return r;
  };
  bf16_t* wqT = (bf16_t*)alloc((size_t)Dm * Dm * 2);
  bf16_t* wkT = (bf16_t*)alloc((size_t)(KVh * HDd) * Dm * 2);
  bf16_t* wvT = (bf16_t*)alloc((size_t)(KVh * HDd) * Dm * 2);
  bf16_t* woT = (bf16_t*)alloc((size_t)Dm * Dm * 2);
  bf16_t* w1T = (bf16_t*)alloc((size_t)Ff * Dm * 2);
  bf16_t* w3T = (bf16_t*)alloc((size_t)Ff * Dm * 2);
  bf16_t* w2T = (bf16_t*)alloc((size_t)Dm * Ff * 2);
  bf16_t* hb  = (bf16_t*)alloc((size_t)Ss * Dm * 2);
  float*  qf  = (float*) alloc((size_t)Ss * Dm * 4);
  float*  kf  = (float*) alloc((size_t)Ss * KVh * HDd * 4);
  float*  vf  = (float*) alloc((size_t)Ss * KVh * HDd * 4);
  bf16_t* qb  = (bf16_t*)alloc((size_t)Ss * Dm * 2);
  bf16_t* kb  = (bf16_t*)alloc((size_t)Ss * KVh * HDd * 2);
  bf16_t* vT  = (bf16_t*)alloc((size_t)KVh * HDd * Ss * 2);
  bf16_t* ob  = (bf16_t*)alloc((size_t)Ss * Dm * 2);
  float*  xm  = (float*) alloc((size_t)Ss * Dm * 4);
  bf16_t* h2b = (bf16_t*)alloc((size_t)Ss * Dm * 2);
  float*  af  = (float*) alloc((size_t)Ss * Ff * 4);
  float*  bf  = (float*) alloc((size_t)Ss * Ff * 4);
  bf16_t* gb  = (bf16_t*)alloc((size_t)Ss * Ff * 2);

  dim3 tb(256);
  // 1) weight transpose+convert (W[K,N] -> WT[N,K] bf16)
  k_transpose_bf16<<<dim3(Dm / 32, Dm / 32), tb, 0, stream>>>(wq, wqT, Dm, Dm);
  k_transpose_bf16<<<dim3((KVh * HDd) / 32, Dm / 32), tb, 0, stream>>>(wk, wkT, Dm, KVh * HDd);
  k_transpose_bf16<<<dim3((KVh * HDd) / 32, Dm / 32), tb, 0, stream>>>(wv, wvT, Dm, KVh * HDd);
  k_transpose_bf16<<<dim3(Dm / 32, Dm / 32), tb, 0, stream>>>(wo, woT, Dm, Dm);
  k_transpose_bf16<<<dim3(Ff / 32, Dm / 32), tb, 0, stream>>>(w1, w1T, Dm, Ff);
  k_transpose_bf16<<<dim3(Ff / 32, Dm / 32), tb, 0, stream>>>(w3, w3T, Dm, Ff);
  k_transpose_bf16<<<dim3(Dm / 32, Ff / 32), tb, 0, stream>>>(w2, w2T, Ff, Dm);

  // 2) h = rmsnorm(x, g_attn)
  k_rmsnorm_bf16<<<dim3(Ss), tb, 0, stream>>>(x, ga, hb);

  auto gemm = [&](const bf16_t* A, const bf16_t* BT, const float* Cin,
                  float* Cout, int M, int N, int K) {
    k_gemm_bf16<<<dim3(N / 128, M / 128), dim3(256), 0, stream>>>(A, BT, Cin, Cout, M, N, K);
  };
  // 3) q/k/v projections
  gemm(hb, wqT, nullptr, qf, Ss, Dm, Dm);
  gemm(hb, wkT, nullptr, kf, Ss, KVh * HDd, Dm);
  gemm(hb, wvT, nullptr, vf, Ss, KVh * HDd, Dm);
  // 4) qk-norm + v transpose
  k_qkv_prep<<<dim3(Ss, Hh + KVh), dim3(128), 0, stream>>>(qf, kf, vf, qb, kb, vT);
  // 5) flash attention
  k_attn<<<dim3(Ss / 16, Hh), dim3(32), 0, stream>>>(qb, kb, vT, ob);
  // 6) x_mid = x + o @ wo
  gemm(ob, woT, x, xm, Ss, Dm, Dm);
  // 7) h2 = rmsnorm(x_mid, g_ffn)
  k_rmsnorm_bf16<<<dim3(Ss), tb, 0, stream>>>(xm, gf, h2b);
  // 8) gate / up
  gemm(h2b, w1T, nullptr, af, Ss, Ff, Dm);
  gemm(h2b, w3T, nullptr, bf, Ss, Ff, Dm);
  // 9) g = silu(a) * b
  size_t n = (size_t)Ss * Ff;
  k_swiglu<<<dim3((unsigned)((n + 255) / 256)), tb, 0, stream>>>(af, bf, gb, n);
  // 10) out = x_mid + g @ w2
  gemm(gb, w2T, xm, out, Ss, Dm, Ff);
}
